// RetrievalModule_13460427505838
// MI455X (gfx1250) — matrix operand order repaired and checked by reading, not compiled, not run on hardware
//
#include <hip/hip_runtime.h>
#include <cstdint>
#include <cstddef>

// ---------------------------------------------------------------------------
// Problem constants (match reference)
// ---------------------------------------------------------------------------
#define D_DIM   768
#define N_BANK  100000
#define B_Q     256
#define K_TOP   5
#define NT_TILES (N_BANK / 16)     // 6250 N-tiles of 16
#define MT_TILES (B_Q / 16)        // 16 M-tiles of 16

typedef __attribute__((ext_vector_type(16))) __bf16 v16bf;
typedef __attribute__((ext_vector_type(8)))  __bf16 v8bf;
typedef __attribute__((ext_vector_type(8)))  float  v8f;
typedef __attribute__((ext_vector_type(2)))  float  v2f;

// ---------------------------------------------------------------------------
// Kernel 1: row L2-normalize (eps-clamped, like F.cosine_similarity) and
// downconvert to bf16.  One wave per row (wave32), 8 rows per 256-thread block.
// ---------------------------------------------------------------------------
__global__ __launch_bounds__(256)
void normalize_bf16_kernel(const float* __restrict__ src, int rows,
                           __bf16* __restrict__ dst) {
    const int lane = threadIdx.x & 31;
    const int wave = threadIdx.x >> 5;
    const int row  = blockIdx.x * 8 + wave;
    if (row >= rows) return;
    const float* x = src + (size_t)row * D_DIM;
    float s = 0.0f;
    #pragma unroll
    for (int j = lane; j < D_DIM; j += 32) { float v = x[j]; s += v * v; }
    #pragma unroll
    for (int off = 16; off > 0; off >>= 1) s += __shfl_xor(s, off, 32);
    const float inv = 1.0f / fmaxf(sqrtf(s), 1e-8f);
    __bf16* y = dst + (size_t)row * D_DIM;
    #pragma unroll
    for (int j = lane; j < D_DIM; j += 32) y[j] = (__bf16)(x[j] * inv);
}

// ---------------------------------------------------------------------------
// Kernel 2: sims[B,N] = qn[B,D] @ cn[N,D]^T  with v_wmma_f32_16x16x32_bf16.
// One wave owns one 16-wide N-tile and ALL 16 M-tiles (B=256) so the bf16
// bank streams from HBM exactly once; qn (0.4MB) re-reads hit L2/WGP$.
//
// Fragment layouts used here (wave32; A and B share contiguous k):
//   A (16x32 bf16): lane m=(l&15); halfs[0..15]=K kb..kb+15, kb=(l>>4)*16.
//   B (32x16 bf16): lane n=(l&15); halfs[0..15]=K kb..kb+15, kb=(l>>4)*16.
//   C/D (16x16 f32): vgpr r, lane l -> M=r+8*(l>>4), N=(l&15).
// ---------------------------------------------------------------------------
__global__ __launch_bounds__(256)
void sims_gemm_kernel(const __bf16* __restrict__ qn,
                      const __bf16* __restrict__ cn,
                      float* __restrict__ sims) {
    const int lane = threadIdx.x & 31;
    const int wave = threadIdx.x >> 5;
    const int nt   = blockIdx.x * 8 + wave;
    if (nt >= NT_TILES) return;                 // wave-uniform guard

    const int lmod = lane & 15;
    const int hi   = lane >> 4;                 // 0 or 1
    const int col  = nt * 16 + lmod;            // this lane's B column

    v8f acc[MT_TILES];
    #pragma unroll
    for (int mt = 0; mt < MT_TILES; ++mt) acc[mt] = (v8f){};

    const __bf16* bbase = cn + (size_t)col * D_DIM + hi * 16;
    const __bf16* abase = qn + (size_t)lmod * D_DIM + hi * 16;

    for (int kk = 0; kk < D_DIM; kk += 32) {
        // B fragment: 32 contiguous bytes of the bank row (two b128 loads)
        v8bf b0 = *(const v8bf*)(bbase + kk);
        v8bf b1 = *(const v8bf*)(bbase + kk + 8);
        v16bf bfrag = __builtin_shufflevector(b0, b1,
            0,1,2,3,4,5,6,7,8,9,10,11,12,13,14,15);

        #pragma unroll
        for (int mt = 0; mt < MT_TILES; ++mt) {
            const __bf16* ap = abase + (size_t)mt * 16 * D_DIM + kk;
            v8bf a0 = *(const v8bf*)(ap);        // K kb .. kb+7
            v8bf a1 = *(const v8bf*)(ap + 8);    // K kb+8 .. kb+15
            v16bf afrag = __builtin_shufflevector(a0, a1,
                0,1,2,3,4,5,6,7,8,9,10,11,12,13,14,15);
            acc[mt] = __builtin_amdgcn_wmma_f32_16x16x32_bf16(
                false, afrag, false, bfrag, (short)0, acc[mt], false, false);
        }
    }

    // Epilogue: scatter accumulators to sims[B, N]
    const int r0 = 8 * hi;
    #pragma unroll
    for (int mt = 0; mt < MT_TILES; ++mt) {
        #pragma unroll
        for (int r = 0; r < 8; ++r) {
            const int row = mt * 16 + r0 + r;
            sims[(size_t)row * N_BANK + col] = acc[mt][r];
        }
    }
}

// ---------------------------------------------------------------------------
// Kernel 3: masked top-5 per query row.  Reference semantics: non-matching
// speakers are filled with -1e30 (still selectable if <5 matches exist);
// has_match = any(same_speaker).  Lowest-index tie-break like lax.top_k.
// ---------------------------------------------------------------------------
__global__ __launch_bounds__(256)
void topk_kernel(const float* __restrict__ sims,
                 const int* __restrict__ speaker_ids,
                 const int* __restrict__ target_ids,
                 int* __restrict__ top_idx,
                 int* __restrict__ has_match) {
    const int b   = blockIdx.x;
    const int tid = threadIdx.x;
    const int tgt = target_ids[b];
    const float* srow = sims + (size_t)b * N_BANK;

    float v[K_TOP];
    int   ix[K_TOP];
    #pragma unroll
    for (int k = 0; k < K_TOP; ++k) { v[k] = -3.0e38f; ix[k] = 0x7fffffff; }

    for (int n = tid; n < N_BANK; n += 256) {
        const float val = (speaker_ids[n] == tgt) ? srow[n] : -1.0e30f;
        if (val > v[K_TOP - 1]) {           // strict > keeps lowest index first
            int k = K_TOP - 1;
            while (k > 0 && v[k - 1] < val) { v[k] = v[k - 1]; ix[k] = ix[k - 1]; --k; }
            v[k] = val; ix[k] = n;
        }
    }

    __shared__ float sv[256];
    __shared__ int   si[256];
    int p = 0;
    for (int k = 0; k < K_TOP; ++k) {
        sv[tid] = (p < K_TOP) ? v[p] : -3.4e38f;
        si[tid] = (p < K_TOP) ? ix[p] : 0x7fffffff;
        __syncthreads();
        for (int s = 128; s > 0; s >>= 1) {
            if (tid < s) {
                const bool take = (sv[tid + s] > sv[tid]) ||
                                  (sv[tid + s] == sv[tid] && si[tid + s] < si[tid]);
                if (take) { sv[tid] = sv[tid + s]; si[tid] = si[tid + s]; }
            }
            __syncthreads();
        }
        const float wv = sv[0];
        const int   wi = si[0];
        if (tid == 0) {
            top_idx[b * K_TOP + k] = wi;
            if (k == 0) has_match[b] = (wv > -1.0e29f) ? 1 : 0;
        }
        __syncthreads();                    // everyone read winner before reuse
        if (p < K_TOP && ix[p] == wi && v[p] == wv) ++p;   // winner advances
    }
}

// ---------------------------------------------------------------------------
// Kernel 4: retrieved = mean of 5 gathered bank rows; combined = [content,
// retrieved] (fp32, exact).
// ---------------------------------------------------------------------------
__global__ __launch_bounds__(256)
void gather_combine_kernel(const float* __restrict__ content,
                           const float* __restrict__ bank,
                           const int* __restrict__ top_idx,
                           float* __restrict__ combined) {
    const int b = blockIdx.x;
    const int* id = top_idx + b * K_TOP;
    float* crow = combined + (size_t)b * (2 * D_DIM);
    const float* xrow = content + (size_t)b * D_DIM;
    for (int j = threadIdx.x; j < D_DIM; j += 256) {
        crow[j] = xrow[j];
        float s = 0.0f;
        #pragma unroll
        for (int k = 0; k < K_TOP; ++k) s += bank[(size_t)id[k] * D_DIM + j];
        crow[D_DIM + j] = s * 0.2f;
    }
}

// ---------------------------------------------------------------------------
// Kernel 5: fp32 GEMM via v_wmma_f32_16x16x4_f32 (keeps the enhance-net
// numerically fp32).  out[M=256, N=768] = act(A[M,K] @ W[K,768] + bias).
// mode 0: SiLU epilogue.  mode 1: has_match select against content.
// A (16x4 f32): lane m=(l&15), vgpr v -> K = (l>>4)*2 + v.
// B (4x16 f32): lane n=(l&15), vgpr v -> K = (l>>4)*2 + v.
// ---------------------------------------------------------------------------
__global__ __launch_bounds__(256)
void mlp_gemm_kernel(const float* __restrict__ A, int Kdim,
                     const float* __restrict__ W,
                     const float* __restrict__ bias,
                     const float* __restrict__ content,
                     const int* __restrict__ has_match,
                     float* __restrict__ out, int mode) {
    const int lane = threadIdx.x & 31;
    const int wave = threadIdx.x >> 5;
    const int tile = blockIdx.x * 8 + wave;           // 16 x 48 = 768 tiles
    if (tile >= MT_TILES * (D_DIM / 16)) return;
    const int mt = tile / (D_DIM / 16);
    const int nt = tile % (D_DIM / 16);

    const int lmod = lane & 15;
    const int hi   = lane >> 4;
    const int row  = mt * 16 + lmod;
    const int col  = nt * 16 + lmod;
    const int kb   = hi * 2;

    v8f acc = (v8f){};
    const float* arow = A + (size_t)row * Kdim + kb;
    const float* wcol = W + col;
    for (int k = 0; k < Kdim; k += 4) {
        v2f a = { arow[k],              arow[k + 1] };
        v2f bb = { wcol[(size_t)(k + kb) * D_DIM], wcol[(size_t)(k + kb + 1) * D_DIM] };
        acc = __builtin_amdgcn_wmma_f32_16x16x4_f32(
            false, a, false, bb, (short)0, acc, false, false);
    }

    const float bv = bias[col];
    const int r0 = 8 * hi;
    #pragma unroll
    for (int r = 0; r < 8; ++r) {
        const int rr = mt * 16 + r0 + r;
        float vv = acc[r] + bv;
        if (mode == 0) {
            vv = vv * (1.0f / (1.0f + __expf(-vv)));              // SiLU
            out[(size_t)rr * D_DIM + col] = vv;
        } else {
            out[(size_t)rr * D_DIM + col] =
                has_match[rr] ? vv : content[(size_t)rr * D_DIM + col];
        }
    }
}

// ---------------------------------------------------------------------------
// Launcher
// ---------------------------------------------------------------------------
extern "C" void kernel_launch(void* const* d_in, const int* in_sizes, int n_in,
                              void* d_out, int out_size, void* d_ws, size_t ws_size,
                              hipStream_t stream) {
    const float* content  = (const float*)d_in[0];
    const int*   tgt_id   = (const int*)  d_in[1];
    const float* bank     = (const float*)d_in[2];
    const int*   spk_id   = (const int*)  d_in[3];
    const float* W1       = (const float*)d_in[4];
    const float* b1       = (const float*)d_in[5];
    const float* W2       = (const float*)d_in[6];
    const float* b2       = (const float*)d_in[7];
    float*       out      = (float*)d_out;

    // Workspace layout (bytes, 16B-aligned slices)
    char* w = (char*)d_ws;
    const size_t off_cn    = 0;                                   // N*D bf16
    const size_t off_qn    = off_cn   + (size_t)N_BANK * D_DIM * 2;
    const size_t off_sims  = off_qn   + (size_t)B_Q * D_DIM * 2;  // B*N f32
    const size_t off_comb  = off_sims + (size_t)B_Q * N_BANK * 4; // B*2D f32
    const size_t off_h     = off_comb + (size_t)B_Q * 2 * D_DIM * 4;
    const size_t off_tidx  = off_h    + (size_t)B_Q * D_DIM * 4;  // B*K int
    const size_t off_match = off_tidx + (size_t)B_Q * K_TOP * 4;  // B int

    __bf16* cn_bf   = (__bf16*)(w + off_cn);
    __bf16* qn_bf   = (__bf16*)(w + off_qn);
    float*  sims    = (float*) (w + off_sims);
    float*  combined= (float*) (w + off_comb);
    float*  h       = (float*) (w + off_h);
    int*    top_idx = (int*)   (w + off_tidx);
    int*    match   = (int*)   (w + off_match);

    // 1) normalize + bf16 downconvert (bank: 12500 blocks, queries: 32)
    normalize_bf16_kernel<<<(N_BANK + 7) / 8, 256, 0, stream>>>(bank, N_BANK, cn_bf);
    normalize_bf16_kernel<<<(B_Q + 7) / 8, 256, 0, stream>>>(content, B_Q, qn_bf);

    // 2) similarity GEMM: 6250 N-tiles, 8 waves/block
    sims_gemm_kernel<<<(NT_TILES + 7) / 8, 256, 0, stream>>>(qn_bf, cn_bf, sims);

    // 3) masked top-5 per query
    topk_kernel<<<B_Q, 256, 0, stream>>>(sims, spk_id, tgt_id, top_idx, match);

    // 4) gather + mean + concat
    gather_combine_kernel<<<B_Q, 256, 0, stream>>>(content, bank, top_idx, combined);

    // 5) enhance net: silu(combined @ W1 + b1) @ W2 + b2, with passthrough
    mlp_gemm_kernel<<<(MT_TILES * (D_DIM / 16)) / 8, 256, 0, stream>>>(
        combined, 2 * D_DIM, W1, b1, nullptr, nullptr, h, 0);
    mlp_gemm_kernel<<<(MT_TILES * (D_DIM / 16)) / 8, 256, 0, stream>>>(
        h, D_DIM, W2, b2, content, match, out, 1);
}
